// GCNsubnet_26353919328686
// MI455X (gfx1250) — matrix operand order repaired and checked by reading, not compiled, not run on hardware
//
#include <hip/hip_runtime.h>
#include <math.h>

// ---------- types ----------
typedef __bf16 bf16_t;
typedef __attribute__((ext_vector_type(16))) __bf16 v16bf;
typedef __attribute__((ext_vector_type(8)))  __bf16 v8bf;
typedef __attribute__((ext_vector_type(8)))  float  v8f;

// ---------- problem constants ----------
#define Bc     32
#define Sc     512
#define Cc     256
#define Gc     256
#define Hc     256
#define Ec     512
#define HEADSc 4
#define HDc    128
#define MROWS  (Bc*Sc)          // 16384

// ---------- WMMA helpers ----------
__device__ __forceinline__ v8f wmma_bf16(v16bf a, v16bf b, v8f c) {
  return __builtin_amdgcn_wmma_f32_16x16x32_bf16(false, a, false, b, (short)0, c, false, false);
}

__device__ __forceinline__ v16bf join_frag(const bf16_t* p0, const bf16_t* p1) {
  v8bf lo = *(const v8bf*)p0;
  v8bf hi = *(const v8bf*)p1;
  v16bf r;
#pragma unroll
  for (int i = 0; i < 8; ++i) { r[i] = lo[i]; r[8 + i] = hi[i]; }
  return r;
}

// C/D layout store: VGPR r, lanes 0-15 -> M=r, N=lane; lanes 16-31 -> M=8+r.
__device__ __forceinline__ void store_acc(const v8f& acc, void* OutP, long outOff, long ldO,
                                          int outBf16, const float* bias, float alpha,
                                          int rowBase, int colBase, int lane) {
  int radd = (lane >> 4) * 8;
  int col  = colBase + (lane & 15);
  float bv = bias ? bias[col] : 0.0f;
#pragma unroll
  for (int r = 0; r < 8; ++r) {
    int row = rowBase + radd + r;
    float v = acc[r] * alpha + bv;
    if (outBf16) ((bf16_t*)OutP)[outOff + (size_t)row * ldO + col] = (bf16_t)v;
    else         ((float*) OutP)[outOff + (size_t)row * ldO + col] = v;
  }
}

// ---------- register-blocked batched GEMM (TN): Out = alpha*A.W^T + bias ----------
// A[M,K] row-major; W[N,K] row-major (both K-contiguous).
// 8 waves/block, each wave computes a 32x32 slab (2x2 WMMA tiles, 4 WMMAs/K-step).
// Block covers 64(M) x 128(N). Requires M%64==0, N%128==0, K%32==0.
__global__ __launch_bounds__(256) void gemm_tn(
    const bf16_t* A, long ldA, long sA1, long sA2,
    const bf16_t* W, long ldW, long sB1, long sB2,
    void* OutP, long ldO, long sO1, long sO2, int outBf16,
    const float* bias, int M, int N, int K, float alpha, int zDiv)
{
  int z = blockIdx.z, z1 = z / zDiv, z2 = z % zDiv;
  A += (size_t)z1 * sA1 + (size_t)z2 * sA2;
  W += (size_t)z1 * sB1 + (size_t)z2 * sB2;
  long outOff = (size_t)z1 * sO1 + (size_t)z2 * sO2;

  int w = threadIdx.x >> 5, lane = threadIdx.x & 31;
  int m0 = (blockIdx.y * 2 + (w >> 2)) * 32;
  int n0 = (blockIdx.x * 4 + (w & 3)) * 32;
  if (m0 >= M || n0 >= N) return;
  int m = lane & 15, kh = lane >> 4;

  const bf16_t* a0 = A + (size_t)(m0 + m)      * ldA;
  const bf16_t* a1 = A + (size_t)(m0 + 16 + m) * ldA;
  const bf16_t* b0 = W + (size_t)(n0 + m)      * ldW;
  const bf16_t* b1 = W + (size_t)(n0 + 16 + m) * ldW;

  v8f acc00 = {}, acc01 = {}, acc10 = {}, acc11 = {};
  for (int k0 = 0; k0 < K; k0 += 32) {
    __builtin_prefetch(a0 + k0 + 128, 0, 1);
    __builtin_prefetch(b0 + k0 + 128, 0, 1);
    v16bf af0 = join_frag(a0 + k0 + kh * 16, a0 + k0 + kh * 16 + 8);
    v16bf af1 = join_frag(a1 + k0 + kh * 16, a1 + k0 + kh * 16 + 8);
    v16bf bf0 = join_frag(b0 + k0 + kh * 16, b0 + k0 + kh * 16 + 8);
    v16bf bf1 = join_frag(b1 + k0 + kh * 16, b1 + k0 + kh * 16 + 8);
    acc00 = wmma_bf16(af0, bf0, acc00);
    acc01 = wmma_bf16(af0, bf1, acc01);
    acc10 = wmma_bf16(af1, bf0, acc10);
    acc11 = wmma_bf16(af1, bf1, acc11);
  }
  store_acc(acc00, OutP, outOff, ldO, outBf16, bias, alpha, m0,      n0,      lane);
  store_acc(acc01, OutP, outOff, ldO, outBf16, bias, alpha, m0,      n0 + 16, lane);
  store_acc(acc10, OutP, outOff, ldO, outBf16, bias, alpha, m0 + 16, n0,      lane);
  store_acc(acc11, OutP, outOff, ldO, outBf16, bias, alpha, m0 + 16, n0 + 16, lane);
}

// ---------- LDS-tiled batched transpose: out[z][c][r] = in[z][r][c] ----------
__global__ __launch_bounds__(256) void transpose_bf(
    const bf16_t* in, long ldIn, long sI1, long sI2,
    bf16_t* out, long ldOut, long sO1, long sO2,
    int R, int Cn, int zDiv)
{
  __shared__ bf16_t tile[32][34];
  int z = blockIdx.z, z1 = z / zDiv, z2 = z % zDiv;
  in  += (size_t)z1 * sI1 + (size_t)z2 * sI2;
  out += (size_t)z1 * sO1 + (size_t)z2 * sO2;
  int c0 = blockIdx.x * 32, r0 = blockIdx.y * 32;
  int tx = threadIdx.x & 31, ty = threadIdx.x >> 5;
#pragma unroll
  for (int i = 0; i < 32; i += 8)
    tile[ty + i][tx] = in[(size_t)(r0 + ty + i) * ldIn + c0 + tx];
  __syncthreads();
#pragma unroll
  for (int i = 0; i < 32; i += 8)
    out[(size_t)(c0 + ty + i) * ldOut + r0 + tx] = tile[tx][ty + i];
}

// ---------- BatchNorm ----------
__global__ void zero_f32(float* p, int n) {
  int i = blockIdx.x * 256 + threadIdx.x;
  if (i < n) p[i] = 0.0f;
}

__global__ void bn_reduce(const float* x, float* sums, float* sqs) {
  int c = threadIdx.x;                       // 256 channels
  size_t r0 = (size_t)blockIdx.x * 256;      // 256 rows per block
  float s = 0.0f, q = 0.0f;
  for (int r = 0; r < 256; ++r) {
    float v = x[(r0 + r) * Cc + c];
    s += v; q += v * v;
  }
  atomicAdd(&sums[c], s);
  atomicAdd(&sqs[c], q);
}

__global__ void bn_apply(const float* x, const float* sums, const float* sqs,
                         const float* gamma, const float* beta, bf16_t* xn) {
  size_t i = (size_t)blockIdx.x * 256 + threadIdx.x;
  int c = (int)(i & (Cc - 1));
  const float invN = 1.0f / (float)MROWS;
  float mu  = sums[c] * invN;
  float var = sqs[c] * invN - mu * mu;
  float rstd = rsqrtf(var + 1e-5f);
  xn[i] = (bf16_t)((x[i] - mu) * rstd * gamma[c] + beta[c]);
}

// ---------- small utilities ----------
__global__ void f2bf(const float* in, bf16_t* out, int n) {
  int i = blockIdx.x * 256 + threadIdx.x;
  if (i < n) out[i] = (bf16_t)in[i];
}

__global__ void add_vec(const float* a, const float* b, float* out, int n) {
  int i = blockIdx.x * 256 + threadIdx.x;
  if (i < n) out[i] = a[i] + b[i];
}

// ---------- BiLSTM (4 blocks: dir x batch-group-of-16; 16 waves each) ----------
__global__ __launch_bounds__(512) void lstm_kernel(
    const bf16_t* xg_f, const bf16_t* xg_b,
    const bf16_t* whh_f, const bf16_t* whh_b,
    float* gatesAll, bf16_t* hcat)
{
  __shared__ bf16_t h_lds[16 * 256];   // h state, WMMA A operand (bf16)
  __shared__ float  c_lds[16 * 256];   // cell state (f32)

  int bx  = blockIdx.x;                // 0..3
  int dir = bx >> 1, bg = bx & 1;
  int b0  = bg * 16;
  const bf16_t* xg  = dir ? xg_b  : xg_f;
  const bf16_t* whh = dir ? whh_b : whh_f;
  float* gates = gatesAll + (size_t)bx * 16 * 1024;

  for (int i = threadIdx.x; i < 16 * 256; i += 512) {
    h_lds[i] = (bf16_t)0.0f;
    c_lds[i] = 0.0f;
  }
  __syncthreads();

  int w = threadIdx.x >> 5, lane = threadIdx.x & 31;
  int m = lane & 15, kh = lane >> 4;

  for (int step = 0; step < Sc; ++step) {
    int t = dir ? (Sc - 1 - step) : step;

    // ---- phase 1: gates_pre = h @ Whh^T  (M=16, N=1024, K=256) ----
    v8f acc[4];
#pragma unroll
    for (int j = 0; j < 4; ++j) acc[j] = (v8f){};
    for (int kk = 0; kk < Gc; kk += 32) {
      const bf16_t* ap = h_lds + (size_t)m * 256 + kk;
      v16bf af = join_frag(ap + kh * 16, ap + kh * 16 + 8);
#pragma unroll
      for (int j = 0; j < 4; ++j) {
        int n = (w * 4 + j) * 16 + m;
        const bf16_t* bp = whh + (size_t)n * Gc + kk + kh * 16;
        v16bf bfr = join_frag(bp, bp + 8);
        acc[j] = wmma_bf16(af, bfr, acc[j]);
      }
    }
#pragma unroll
    for (int j = 0; j < 4; ++j) {
#pragma unroll
      for (int r = 0; r < 8; ++r) {
        int row = r + (lane >> 4) * 8;            // batch in group (0..15)
        int col = (w * 4 + j) * 16 + (lane & 15); // gate column (0..1023)
        float xv = (float)xg[((size_t)(b0 + row) * Sc + t) * 1024 + col];
        gates[row * 1024 + col] = acc[j][r] + xv;
      }
    }
    __syncthreads();

    // ---- phase 2: nonlinearity + state update ----
    for (int u = threadIdx.x; u < 16 * 256; u += 512) {
      int b = u >> 8, j = u & 255;
      float ig = gates[b * 1024 + j];
      float fg = gates[b * 1024 + 256 + j];
      float gg = gates[b * 1024 + 512 + j];
      float og = gates[b * 1024 + 768 + j];
      float c  = c_lds[u];
      float si = 1.0f / (1.0f + __expf(-ig));
      float sf = 1.0f / (1.0f + __expf(-fg));
      float so = 1.0f / (1.0f + __expf(-og));
      float tg = tanhf(gg);
      c = sf * c + si * tg;
      float h = so * tanhf(c);
      c_lds[u] = c;
      h_lds[u] = (bf16_t)h;
      hcat[((size_t)(b0 + b) * Sc + t) * 512 + dir * 256 + j] = (bf16_t)h;
    }
    __syncthreads();
  }
}

// ---------- softmax over rows of 512 (one block per row) ----------
__global__ __launch_bounds__(256) void softmax_kernel(const float* scores, bf16_t* att) {
  __shared__ float red[256];
  size_t row = blockIdx.x;
  const float* p = scores + row * 512;
  int t = threadIdx.x;
  float v0 = p[t], v1 = p[t + 256];
  red[t] = fmaxf(v0, v1);
  __syncthreads();
  for (int s = 128; s > 0; s >>= 1) { if (t < s) red[t] = fmaxf(red[t], red[t + s]); __syncthreads(); }
  float rowmax = red[0];
  __syncthreads();
  float e0 = __expf(v0 - rowmax), e1 = __expf(v1 - rowmax);
  red[t] = e0 + e1;
  __syncthreads();
  for (int s = 128; s > 0; s >>= 1) { if (t < s) red[t] += red[t + s]; __syncthreads(); }
  float inv = 1.0f / red[0];
  att[row * 512 + t]       = (bf16_t)(e0 * inv);
  att[row * 512 + t + 256] = (bf16_t)(e1 * inv);
}

// ---------- graph construction ----------
__global__ void sw_kernel(const bf16_t* att, float* Aadj) {
  size_t idx = (size_t)blockIdx.x * 256 + threadIdx.x;   // over B*S*S
  size_t b = idx >> 18;
  size_t r = idx & 262143;
  int i = (int)(r >> 9), j = (int)(r & 511);
  float s = 0.0f;
#pragma unroll
  for (int h = 0; h < 4; ++h) s += (float)att[(b * 4 + h) * 262144 + r];
  float v = s * 0.25f;
  v = v > 0.0f ? v : 0.0f;
  if (i == j && v == 0.0f) v = 1.0f;   // add_remaining_self_loops
  Aadj[idx] = v;
}

__global__ void deg_kernel(const float* Aadj, float* dinv) {
  int b = blockIdx.x >> 1;
  int j = ((blockIdx.x & 1) << 8) + threadIdx.x;
  const float* p = Aadj + (size_t)b * 262144;
  float s = 0.0f;
  for (int i = 0; i < 512; ++i) s += p[(size_t)i * 512 + j];
  dinv[b * 512 + j] = s > 0.0f ? rsqrtf(s) : 0.0f;
}

// Store An transposed (AnT[b,j,i]) so the aggregation GEMM's A operand is K-contiguous.
__global__ void ant_kernel(const float* Aadj, const float* dinv, bf16_t* AnT) {
  size_t idx = (size_t)blockIdx.x * 256 + threadIdx.x;   // over B*S*S
  size_t b = idx >> 18;
  size_t r = idx & 262143;
  int j = (int)(r >> 9), i = (int)(r & 511);
  float v = Aadj[b * 262144 + (size_t)i * 512 + j] * dinv[b * 512 + i] * dinv[b * 512 + j];
  AnT[idx] = (bf16_t)v;
}

// ---------- host-side launcher ----------
extern "C" void kernel_launch(void* const* d_in, const int* in_sizes, int n_in,
                              void* d_out, int out_size, void* d_ws, size_t ws_size,
                              hipStream_t stream) {
  const float* x         = (const float*)d_in[0];
  const float* bn_gamma  = (const float*)d_in[1];
  const float* bn_beta   = (const float*)d_in[2];
  const float* w_ih_f    = (const float*)d_in[3];
  const float* w_hh_f    = (const float*)d_in[4];
  const float* b_ih_f    = (const float*)d_in[5];
  const float* b_hh_f    = (const float*)d_in[6];
  const float* w_ih_b    = (const float*)d_in[7];
  const float* w_hh_b    = (const float*)d_in[8];
  const float* b_ih_b    = (const float*)d_in[9];
  const float* b_hh_b    = (const float*)d_in[10];
  const float* glin_w    = (const float*)d_in[11];
  const float* glin_b    = (const float*)d_in[12];
  const float* elin_w    = (const float*)d_in[13];
  const float* elin_b    = (const float*)d_in[14];
  const float* in_proj_w = (const float*)d_in[15];
  const float* in_proj_b = (const float*)d_in[16];
  const float* out_proj_w= (const float*)d_in[17];
  const float* out_proj_b= (const float*)d_in[18];
  const float* gcn1_w    = (const float*)d_in[19];
  const float* gcn1_b    = (const float*)d_in[20];
  const float* gcn2_w    = (const float*)d_in[21];
  const float* gcn2_b    = (const float*)d_in[22];

  char* ws = (char*)d_ws;
  size_t off = 0;
  auto alloc = [&](size_t bytes) -> void* {
    void* p = ws + off;
    off += (bytes + 255) & ~(size_t)255;
    return p;
  };

  float*  sums      = (float*)alloc(2 * 256 * sizeof(float));
  float*  sqs       = sums + 256;
  bf16_t* xn_bf     = (bf16_t*)alloc((size_t)MROWS * Cc * 2);
  bf16_t* wihf_bf   = (bf16_t*)alloc(1024 * 256 * 2);
  bf16_t* whhf_bf   = (bf16_t*)alloc(1024 * 256 * 2);
  bf16_t* wihb_bf   = (bf16_t*)alloc(1024 * 256 * 2);
  bf16_t* whhb_bf   = (bf16_t*)alloc(1024 * 256 * 2);
  bf16_t* glin_bf   = (bf16_t*)alloc(256 * 512 * 2);
  bf16_t* elin_bf   = (bf16_t*)alloc(256 * 256 * 2);
  bf16_t* inproj_bf = (bf16_t*)alloc(1536 * 512 * 2);
  bf16_t* outproj_bf= (bf16_t*)alloc(512 * 512 * 2);
  bf16_t* gcn1_bf   = (bf16_t*)alloc(512 * 256 * 2);
  bf16_t* gcn2_bf   = (bf16_t*)alloc(256 * 512 * 2);
  float*  bias_f    = (float*)alloc(1024 * sizeof(float));
  float*  bias_b    = (float*)alloc(1024 * sizeof(float));
  bf16_t* xg_f      = (bf16_t*)alloc((size_t)MROWS * 1024 * 2);
  bf16_t* xg_b      = (bf16_t*)alloc((size_t)MROWS * 1024 * 2);
  float*  gates     = (float*)alloc(4 * 16 * 1024 * sizeof(float));
  bf16_t* hcat_bf   = (bf16_t*)alloc((size_t)MROWS * 512 * 2);
  bf16_t* x1_bf     = (bf16_t*)alloc((size_t)MROWS * 512 * 2);
  bf16_t* qkv_bf    = (bf16_t*)alloc((size_t)MROWS * 1536 * 2);
  float*  scores    = (float*)alloc((size_t)Bc * HEADSc * Sc * Sc * sizeof(float));
  bf16_t* att_bf    = (bf16_t*)alloc((size_t)Bc * HEADSc * Sc * Sc * 2);
  bf16_t* vT_bf     = (bf16_t*)alloc((size_t)Bc * HEADSc * HDc * Sc * 2);
  bf16_t* lu_pre    = (bf16_t*)alloc((size_t)MROWS * 512 * 2);
  float*  Aadj      = (float*)alloc((size_t)Bc * Sc * Sc * sizeof(float));
  float*  dinv      = (float*)alloc((size_t)Bc * Sc * sizeof(float));
  bf16_t* AnT_bf    = (bf16_t*)alloc((size_t)Bc * Sc * Sc * 2);
  bf16_t* hw1_bf    = (bf16_t*)alloc((size_t)MROWS * 512 * 2);
  bf16_t* hw1T_bf   = (bf16_t*)alloc((size_t)MROWS * 512 * 2);
  bf16_t* h1_bf     = (bf16_t*)alloc((size_t)MROWS * 512 * 2);
  bf16_t* hw2_bf    = (bf16_t*)alloc((size_t)MROWS * 256 * 2);
  bf16_t* hw2T_bf   = (bf16_t*)alloc((size_t)MROWS * 256 * 2);

  float* out_gcn = (float*)d_out;                         // [B,S,H]
  float* out_lu  = (float*)d_out + (size_t)Bc * Sc * Hc;  // [B,S,E]

  auto launch_tn = [&](const bf16_t* A, long ldA, long sA1, long sA2,
                       const bf16_t* W, long ldW, long sB1, long sB2,
                       void* Out, long ldO, long sO1, long sO2, int outBf,
                       const float* bias, int M, int N, int K, float alpha,
                       int zDiv, int Z) {
    dim3 g(N / 128, M / 64, Z);
    gemm_tn<<<g, 256, 0, stream>>>(A, ldA, sA1, sA2, W, ldW, sB1, sB2,
                                   Out, ldO, sO1, sO2, outBf, bias, M, N, K, alpha, zDiv);
  };
  auto launch_tr = [&](const bf16_t* in, long ldIn, long sI1, long sI2,
                       bf16_t* outp, long ldOut, long sO1, long sO2,
                       int R, int Cn, int zDiv, int Z) {
    dim3 g(Cn / 32, R / 32, Z);
    transpose_bf<<<g, 256, 0, stream>>>(in, ldIn, sI1, sI2, outp, ldOut, sO1, sO2, R, Cn, zDiv);
  };
  auto conv = [&](const float* in, bf16_t* outp, int n) {
    f2bf<<<(n + 255) / 256, 256, 0, stream>>>(in, outp, n);
  };

  // 1) BatchNorm (training-mode batch stats)
  zero_f32<<<2, 256, 0, stream>>>(sums, 512);
  bn_reduce<<<MROWS / 256, 256, 0, stream>>>(x, sums, sqs);
  bn_apply<<<(MROWS * Cc) / 256, 256, 0, stream>>>(x, sums, sqs, bn_gamma, bn_beta, xn_bf);

  // 2) weight conversions + fused LSTM biases
  conv(w_ih_f, wihf_bf, 1024 * 256);
  conv(w_hh_f, whhf_bf, 1024 * 256);
  conv(w_ih_b, wihb_bf, 1024 * 256);
  conv(w_hh_b, whhb_bf, 1024 * 256);
  conv(glin_w, glin_bf, 256 * 512);
  conv(elin_w, elin_bf, 256 * 256);
  conv(in_proj_w, inproj_bf, 1536 * 512);
  conv(out_proj_w, outproj_bf, 512 * 512);
  conv(gcn1_w, gcn1_bf, 512 * 256);
  conv(gcn2_w, gcn2_bf, 256 * 512);
  add_vec<<<4, 256, 0, stream>>>(b_ih_f, b_hh_f, bias_f, 1024);
  add_vec<<<4, 256, 0, stream>>>(b_ih_b, b_hh_b, bias_b, 1024);

  // 3) LSTM input projections: xg = xn @ w_ih^T + (b_ih + b_hh)
  launch_tn(xn_bf, Cc, 0, 0, wihf_bf, Cc, 0, 0, xg_f, 1024, 0, 0, 1, bias_f,
            MROWS, 1024, Cc, 1.0f, 1, 1);
  launch_tn(xn_bf, Cc, 0, 0, wihb_bf, Cc, 0, 0, xg_b, 1024, 0, 0, 1, bias_b,
            MROWS, 1024, Cc, 1.0f, 1, 1);

  // 4) BiLSTM recurrence (WMMA recurrent GEMM per step)
  lstm_kernel<<<4, 512, 0, stream>>>(xg_f, xg_b, whhf_bf, whhb_bf, gates, hcat_bf);

  // 5) g = hcat @ glin^T + b  -> x1[:,256:512] ; elin(xn) -> x1[:,0:256]
  launch_tn(hcat_bf, 512, 0, 0, glin_bf, 512, 0, 0, x1_bf + 256, 512, 0, 0, 1,
            glin_b, MROWS, Gc, 512, 1.0f, 1, 1);
  launch_tn(xn_bf, Cc, 0, 0, elin_bf, Cc, 0, 0, x1_bf, 512, 0, 0, 1,
            elin_b, MROWS, Gc, Cc, 1.0f, 1, 1);

  // 6) qkv = x1 @ in_proj^T + b
  launch_tn(x1_bf, Ec, 0, 0, inproj_bf, Ec, 0, 0, qkv_bf, 3 * Ec, 0, 0, 1,
            in_proj_b, MROWS, 3 * Ec, Ec, 1.0f, 1, 1);

  // 7) scores[b,h] = q k^T / sqrt(hd)   (batched over B*HEADS)
  launch_tn(qkv_bf, 3 * Ec, (long)Sc * 3 * Ec, HDc,
            qkv_bf + Ec, 3 * Ec, (long)Sc * 3 * Ec, HDc,
            scores, Sc, (long)HEADSc * Sc * Sc, (long)Sc * Sc, 0,
            nullptr, Sc, Sc, HDc, 1.0f / sqrtf((float)HDc), HEADSc, Bc * HEADSc);

  // 8) softmax rows -> att (bf16)
  softmax_kernel<<<Bc * HEADSc * Sc, 256, 0, stream>>>(scores, att_bf);

  // 9) V^T per (b,h), then lu_heads = att @ V as TN GEMM
  launch_tr(qkv_bf + 2 * Ec, 3 * Ec, (long)Sc * 3 * Ec, HDc,
            vT_bf, Sc, (long)HEADSc * HDc * Sc, (long)HDc * Sc,
            Sc, HDc, HEADSc, Bc * HEADSc);
  launch_tn(att_bf, Sc, (long)HEADSc * Sc * Sc, (long)Sc * Sc,
            vT_bf, Sc, (long)HEADSc * HDc * Sc, (long)HDc * Sc,
            lu_pre, Ec, (long)Sc * Ec, HDc, 1,
            nullptr, Sc, HDc, Sc, 1.0f, HEADSc, Bc * HEADSc);

  // 10) lu = lu_pre @ out_proj^T + b  -> d_out (second region, f32)
  launch_tn(lu_pre, Ec, 0, 0, outproj_bf, Ec, 0, 0, out_lu, Ec, 0, 0, 0,
            out_proj_b, MROWS, Ec, Ec, 1.0f, 1, 1);

  // 11) graph: Sw = relu(mean_h att) + self loops; deg^-1/2; An (stored transposed)
  sw_kernel<<<(Bc * Sc * Sc) / 256, 256, 0, stream>>>(att_bf, Aadj);
  deg_kernel<<<Bc * 2, 256, 0, stream>>>(Aadj, dinv);
  ant_kernel<<<(Bc * Sc * Sc) / 256, 256, 0, stream>>>(Aadj, dinv, AnT_bf);

  // 12) GCN layer 1: hw1 = xn @ gcn1^T; transpose; h1 = An^T hw1 + b1 (TN)
  launch_tn(xn_bf, Cc, 0, 0, gcn1_bf, Cc, 0, 0, hw1_bf, 2 * Hc, 0, 0, 1,
            nullptr, MROWS, 2 * Hc, Cc, 1.0f, 1, 1);
  launch_tr(hw1_bf, 2 * Hc, (long)Sc * 2 * Hc, 0,
            hw1T_bf, Sc, (long)Sc * 2 * Hc, 0,
            Sc, 2 * Hc, 1, Bc);
  launch_tn(AnT_bf, Sc, (long)Sc * Sc, 0,
            hw1T_bf, Sc, (long)Sc * 2 * Hc, 0,
            h1_bf, 2 * Hc, (long)Sc * 2 * Hc, 0, 1,
            gcn1_b, Sc, 2 * Hc, Sc, 1.0f, 1, Bc);

  // 13) GCN layer 2: hw2 = h1 @ gcn2^T; transpose; out = An^T hw2 + b2 -> d_out
  launch_tn(h1_bf, 2 * Hc, 0, 0, gcn2_bf, 2 * Hc, 0, 0, hw2_bf, Hc, 0, 0, 1,
            nullptr, MROWS, Hc, 2 * Hc, 1.0f, 1, 1);
  launch_tr(hw2_bf, Hc, (long)Sc * Hc, 0,
            hw2T_bf, Sc, (long)Sc * Hc, 0,
            Sc, Hc, 1, Bc);
  launch_tn(AnT_bf, Sc, (long)Sc * Sc, 0,
            hw2T_bf, Sc, (long)Sc * Hc, 0,
            out_gcn, Hc, (long)Sc * Hc, 0, 0,
            gcn2_b, Sc, Hc, Sc, 1.0f, 1, Bc);

  (void)in_sizes; (void)n_in; (void)out_size; (void)ws_size;
}
